// PCELayer_59416577573253
// MI455X (gfx1250) — matrix operand order repaired and checked by reading, not compiled, not run on hardware
//
#include <hip/hip_runtime.h>
#include <hip/hip_bf16.h>
#include <stdint.h>

// ---------------------------------------------------------------------------
// MoE patch-gated 3x3 conv experts for MI455X (gfx1250, wave32, WMMA).
//   1) prep_weights : We [E,O,C,3,3] f32 -> wbuf [E,9,O,C] bf16
//   2) prep_input   : x  [B,C,H,W]  f32 -> xpad [B,130,130,C] bf16 (zero halo)
//   3) gate_kernel  : patch-pool + fourier + MLP + softmax -> gates [512,8] f32
//   4) moe_conv     : per-patch implicit GEMM with V_WMMA_F32_16X16X32_BF16.
//                     Halo tile staged once; per-tap 8KB weight blocks stream
//                     through a double-buffered LDS ping-pong via
//                     GLOBAL_LOAD_ASYNC_TO_LDS_B128 (ASYNCcnt), so the DMA of
//                     tap t+1 overlaps the 16 WMMAs of tap t.
// ---------------------------------------------------------------------------

typedef __attribute__((ext_vector_type(16))) __bf16 v16bf;
typedef __attribute__((ext_vector_type(8)))  __bf16 v8bf;
typedef __attribute__((ext_vector_type(8)))  float  v8f;

#define B_    8
#define C_    64
#define H_    128
#define W_    128
#define E_    8
#define O_    64
#define HID_  128
#define GC_   96
#define P_    16
#define PAD_H 130
#define PAD_W 130
#define TAPW  (O_ * C_)          // 4096 bf16 = 8KB per (expert,tap) block
#define NSTG  (E_ * 9)           // 72 pipeline stages

static __device__ __forceinline__ __bf16 f2bf(float f) {
  unsigned u = __builtin_bit_cast(unsigned, f);
  unsigned r = (u + 0x7FFFu + ((u >> 16) & 1u)) >> 16;   // round-to-nearest-even
  unsigned short s = (unsigned short)r;
  return __builtin_bit_cast(__bf16, s);
}

// LDS byte offset of a __shared__ object (flat LDS aperture: low 32 bits).
static __device__ __forceinline__ unsigned lds_off(const void* p) {
  return (unsigned)(uintptr_t)p;
}

// Async DMA: 16 bytes global -> LDS, tracked by ASYNCcnt (per-lane addresses).
static __device__ __forceinline__ void async_copy_b128(unsigned lds_byte_off,
                                                       const void* gsrc) {
  asm volatile("global_load_async_to_lds_b128 %0, %1, off"
               :: "v"(lds_byte_off), "v"(gsrc) : "memory");
}
static __device__ __forceinline__ void wait_async0() {
  asm volatile("s_wait_asynccnt 0x0" ::: "memory");
}

static __device__ __forceinline__ float relu_med3(float v) {
  return __builtin_amdgcn_fmed3f(v, 0.0f, 3.402823466e+38f);  // v_med3_num_f32
}

// ---- 1) weights: We[e][o][c][dy][dx] f32 -> wbuf[((e*9+tap)*O+o)*C+c] bf16 --
__global__ __launch_bounds__(256) void prep_weights(const float* __restrict__ We,
                                                    __bf16* __restrict__ wbuf) {
  int id = blockIdx.x * 256 + threadIdx.x;
  if (id >= E_ * 9 * O_ * C_) return;
  int c   = id & 63;
  int o   = (id >> 6) & 63;
  int tap = (id >> 12) % 9;
  int e   = id / (9 * 64 * 64);
  int dy = tap / 3, dx = tap - dy * 3;
  float w = We[(((size_t)(e * O_ + o) * C_ + c) * 3 + dy) * 3 + dx];
  wbuf[id] = f2bf(w);
}

// ---- 2) input: NCHW f32 -> padded NHWC bf16 (1-pixel zero halo) -------------
__global__ __launch_bounds__(64) void prep_input(const float* __restrict__ x,
                                                 __bf16* __restrict__ xpad) {
  int i = blockIdx.x;                 // b*130*130 + y*130 + xx
  int c = threadIdx.x;                // 0..63
  int b = i / (PAD_H * PAD_W);
  int rem = i - b * PAD_H * PAD_W;
  int y  = rem / PAD_W;
  int xx = rem - y * PAD_W;
  float v = 0.f;
  if (y >= 1 && y <= H_ && xx >= 1 && xx <= W_)
    v = x[((size_t)(b * C_ + c) * H_ + (y - 1)) * W_ + (xx - 1)];
  xpad[(size_t)i * C_ + c] = f2bf(v);
}

// ---- 3) router: pooled channels + fourier feats -> MLP -> softmax gates -----
__global__ __launch_bounds__(128) void gate_kernel(const float* __restrict__ x,
                                                   const float* __restrict__ W1,
                                                   const float* __restrict__ b1,
                                                   const float* __restrict__ W2,
                                                   const float* __restrict__ b2,
                                                   float* __restrict__ gates) {
  __shared__ float g[GC_];
  __shared__ float h[HID_];
  __shared__ float part[128];
  __shared__ float lg[E_];

  const int p  = blockIdx.x;          // 0..511
  const int b  = p >> 6;
  const int ph = (p >> 3) & 7;
  const int pw = p & 7;
  const int tid = threadIdx.x;

  const int c  = tid & 63;
  const int hf = tid >> 6;
  float s = 0.f;
  for (int ry = 0; ry < 8; ++ry) {
    const int y = ph * P_ + hf * 8 + ry;
    const float* row = x + ((size_t)(b * C_ + c) * H_ + y) * W_ + pw * P_;
    #pragma unroll
    for (int cx = 0; cx < 16; ++cx) s += row[cx];
  }
  part[tid] = s;
  __syncthreads();
  if (tid < 64) g[tid] = (part[tid] + part[tid + 64]) * (1.f / 256.f);
  if (tid >= 64 && tid < 96) {
    const int t2 = tid - 64;
    const int f = t2 & 7, kind = t2 >> 3;   // 0:sin(u) 1:cos(u) 2:sin(v) 3:cos(v)
    const float u = (ph + 0.5f) / 8.f;
    const float v = (pw + 0.5f) / 8.f;
    const float ang = 6.283185307179586f * (float)(1 << f) * ((kind < 2) ? u : v);
    g[64 + kind * 8 + f] = (kind & 1) ? cosf(ang) : sinf(ang);
  }
  __syncthreads();

  float acc = b1[tid];
  for (int cc = 0; cc < GC_; ++cc) acc += W1[tid * GC_ + cc] * g[cc];
  h[tid] = relu_med3(acc);
  __syncthreads();

  if (tid < E_) {
    float a = b2[tid];
    for (int j = 0; j < HID_; ++j) a += W2[tid * HID_ + j] * h[j];
    lg[tid] = a;
  }
  __syncthreads();

  if (tid == 0) {
    float mx = lg[0];
    for (int e = 1; e < E_; ++e) mx = fmaxf(mx, lg[e]);
    float ex[E_], sum = 0.f;
    for (int e = 0; e < E_; ++e) { ex[e] = expf(lg[e] - mx); sum += ex[e]; }
    const float inv = 1.f / sum;
    for (int e = 0; e < E_; ++e) gates[p * E_ + e] = ex[e] * inv;
  }
}

// ---- 4) conv experts + gated combine ---------------------------------------
// One block per 16x16 patch (512 blocks), 8 waves; wave w owns output rows
// {2w,2w+1}. M=16 pixels, N=4x16 out channels, K=9 taps x 64 channels.
// LDS: halo 41472 B + 2 x 8192 B weight ping-pong = 57856 B  (~5 blocks/WGP).
// Pipeline stage s = (e*9+tap): prefetch stage s+1 weights via async LDS-DMA,
// run 16 WMMAs of stage s, s_wait_asynccnt 0, barrier, swap buffers.
__global__ __launch_bounds__(256) void moe_conv(const __bf16* __restrict__ xpad,
                                                const __bf16* __restrict__ wbuf,
                                                const float* __restrict__ be,
                                                const float* __restrict__ gates,
                                                float* __restrict__ out) {
  __shared__ __align__(16) __bf16 ldsx[18 * 18 * C_];   // 41472 B halo tile
  __shared__ __align__(16) __bf16 ldsw[2][TAPW];        // 2 x 8192 B tap weights

  const int p  = blockIdx.x;
  const int b  = p >> 6;
  const int ph = (p >> 3) & 7;
  const int pw = p & 7;
  const int y0 = ph * P_, x0 = pw * P_;
  const int tid  = threadIdx.x;
  const int wave = tid >> 5;
  const int lane = tid & 31;
  const int m    = lane & 15;     // M / N index within a 16-wide tile
  const int half = lane >> 4;     // lane group (selects K sub-range)

  const unsigned ldsx_base = lds_off(ldsx);
  const unsigned ldsw_base = lds_off(ldsw);

  // Async halo staging: 18x18 sites x 64 ch bf16, 16B chunks.
  {
    const size_t gbase = ((size_t)(b * PAD_H + y0) * PAD_W + x0) * C_;
    for (int idx = tid; idx < 18 * 18 * 8; idx += 256) {
      const int site = idx >> 3;
      const int q    = idx & 7;
      const int row  = site / 18;
      const int col  = site - row * 18;
      async_copy_b128(ldsx_base + (unsigned)(((row * 18 + col) * C_ + q * 8) * 2),
                      xpad + gbase + ((size_t)row * PAD_W + col) * C_ + q * 8);
    }
  }
  // Stage 0 weights into buffer 0 (512 x 16B chunks, 2 per thread).
  for (int idx = tid; idx < TAPW / 8; idx += 256)
    async_copy_b128(ldsw_base + (unsigned)(idx * 16), wbuf + idx * 8);
  wait_async0();
  __syncthreads();

  v8f accf[2][4];
  #pragma unroll
  for (int rr = 0; rr < 2; ++rr)
    #pragma unroll
    for (int t = 0; t < 4; ++t)
      #pragma unroll
      for (int j = 0; j < 8; ++j) accf[rr][t][j] = 0.f;

  for (int e = 0; e < E_; ++e) {
    v8f acc[2][4];
    #pragma unroll
    for (int t = 0; t < 4; ++t) {
      const float bv = be[e * O_ + t * 16 + m];   // col o = t*16 + (lane&15)
      #pragma unroll
      for (int j = 0; j < 8; ++j) { acc[0][t][j] = bv; acc[1][t][j] = bv; }
    }

    for (int tap = 0; tap < 9; ++tap) {
      const int s   = e * 9 + tap;
      const int buf = s & 1;

      // Prefetch next tap's 8KB weight block into the other buffer.
      if (s + 1 < NSTG) {
        const __bf16* wsrc = wbuf + (size_t)(s + 1) * TAPW;
        const unsigned dstb = ldsw_base + (unsigned)(((s + 1) & 1) * (TAPW * 2));
        for (int idx = tid; idx < TAPW / 8; idx += 256)
          async_copy_b128(dstb + (unsigned)(idx * 16), wsrc + idx * 8);
      }

      const int dy = tap / 3, dx = tap - dy * 3;
      const __bf16* wl = &ldsw[buf][0];
      #pragma unroll
      for (int k0 = 0; k0 < C_; k0 += 32) {
        // B frags from LDS: lane<16 -> col n, K k0..k0+15 ; lane>=16 -> +16
        v16bf bf[4];
        #pragma unroll
        for (int t = 0; t < 4; ++t) {
          const int n = t * 16 + m;
          bf[t] = *(const v16bf*)(wl + n * C_ + k0 + half * 16);
        }
        // A frags from LDS, rows r = 2*wave + rr; then 8 WMMAs
        #pragma unroll
        for (int rr = 0; rr < 2; ++rr) {
          const int r = wave * 2 + rr;
          const int base = ((r + dy) * 18 + (m + dx)) * C_ + k0 + half * 16;
          union { v16bf v; v8bf h[2]; } A;
          A.h[0] = *(const v8bf*)(ldsx + base);        // K k0+{0..7}  (+16 if half)
          A.h[1] = *(const v8bf*)(ldsx + base + 8);    // K k0+{8..15} (+16 if half)
          #pragma unroll
          for (int t = 0; t < 4; ++t)
            acc[rr][t] = __builtin_amdgcn_wmma_f32_16x16x32_bf16(
                false, A.v, false, bf[t], (short)0, acc[rr][t], false, false);
        }
      }

      // Close the stage: my prefetch landed; all waves done reading buf.
      wait_async0();
      __syncthreads();
    }

    // ReLU (v_med3), gate (uniform per patch), fold into final accumulator.
    const float ge = gates[p * E_ + e];
    #pragma unroll
    for (int rr = 0; rr < 2; ++rr)
      #pragma unroll
      for (int t = 0; t < 4; ++t)
        #pragma unroll
        for (int j = 0; j < 8; ++j)
          accf[rr][t][j] += ge * relu_med3(acc[rr][t][j]);
  }

  // Store: D element (vgpr j, lane) -> x = x0 + j + half*8, o = t*16 + m.
  #pragma unroll
  for (int rr = 0; rr < 2; ++rr) {
    const int y = y0 + wave * 2 + rr;
    #pragma unroll
    for (int t = 0; t < 4; ++t) {
      const int o = t * 16 + m;
      float* outp = out + ((size_t)(b * O_ + o) * H_ + y) * W_ + x0 + half * 8;
      #pragma unroll
      for (int j = 0; j < 8; ++j) outp[j] = accf[rr][t][j];
    }
  }
}

// ---------------------------------------------------------------------------
extern "C" void kernel_launch(void* const* d_in, const int* in_sizes, int n_in,
                              void* d_out, int out_size, void* d_ws, size_t ws_size,
                              hipStream_t stream) {
  const float* x  = (const float*)d_in[0];
  const float* We = (const float*)d_in[1];
  const float* be = (const float*)d_in[2];
  const float* W1 = (const float*)d_in[3];
  const float* b1 = (const float*)d_in[4];
  const float* W2 = (const float*)d_in[5];
  const float* b2 = (const float*)d_in[6];
  float* out = (float*)d_out;

  char* ws = (char*)d_ws;
  float*  gates = (float*)ws;                                   //   16,384 B
  __bf16* wbuf  = (__bf16*)(ws + 16384);                        //  589,824 B
  __bf16* xpad  = (__bf16*)(ws + 16384 + 589824);               // 17,305,600 B

  prep_weights<<<(E_ * 9 * O_ * C_ + 255) / 256, 256, 0, stream>>>(We, wbuf);
  prep_input<<<B_ * PAD_H * PAD_W, 64, 0, stream>>>(x, xpad);
  gate_kernel<<<512, 128, 0, stream>>>(x, W1, b1, W2, b2, gates);
  moe_conv<<<512, 256, 0, stream>>>(xpad, wbuf, be, gates, out);
}
